// MultiHeadAttention_26955214750378
// MI455X (gfx1250) — matrix operand order repaired and checked by reading, not compiled, not run on hardware
//
#include <hip/hip_runtime.h>

// ---------------------------------------------------------------------------
// MHA forward for gfx1250 (MI455X): bf16 WMMA GEMM pipeline with
// double-buffered async global->LDS tile loads (ASYNCcnt path).
//   d_out layout: [ out: 2*2048*1024 f32 | attn: 2*16*2048*2048 f32 ]
//   workspace (~48 MB): bf16 copies of x / packed Wqkv / Wo, Q/K head-major,
//   V transposed per head, ctx bf16.
// ---------------------------------------------------------------------------

typedef __bf16 v16bf __attribute__((ext_vector_type(16)));
typedef float  v8f   __attribute__((ext_vector_type(8)));
typedef int    v4i   __attribute__((ext_vector_type(4)));

typedef __attribute__((address_space(1))) v4i* gptr_b128;   // global int4*
typedef __attribute__((address_space(3))) v4i* lptr_b128;   // LDS int4*

#define D_MODEL   1024
#define N_HEADS   16
#define HEAD_DIM  64
#define BATCH     2
#define SEQ       2048
#define MTOT      (BATCH * SEQ)        // 4096 tokens

// workgroup GEMM tile
#define BM 128
#define BN 128
#define BK 32

#if __has_builtin(__builtin_amdgcn_global_load_async_to_lds_b128)
#define HAS_ASYNC_LDS 1
#else
#define HAS_ASYNC_LDS 0
#endif

__device__ __forceinline__ void wait_async_lds() {
#if __has_builtin(__builtin_amdgcn_s_wait_asynccnt)
  __builtin_amdgcn_s_wait_asynccnt(0);
#else
  asm volatile("s_wait_asynccnt 0x0" ::: "memory");
#endif
}

__device__ __forceinline__ unsigned short f2bf(float f) {
  unsigned int u = __float_as_uint(f);
  u += 0x7FFFu + ((u >> 16) & 1u);     // round-to-nearest-even
  return (unsigned short)(u >> 16);
}

// ---- LDS fragment gathers (CDNA5 16-bit WMMA operand layouts, ISA 7.12.2) --
// A 16x32: lane<16 holds K{0..7}+{16..23}, lane>=16 holds K{8..15}+{24..31}
__device__ __forceinline__ v16bf load_a_frag(const unsigned short* As, int row, int half) {
  v16bf a;
  const unsigned short* p = As + row * BK;
  *((uint4*)&a)       = *(const uint4*)(p + half * 8);
  *(((uint4*)&a) + 1) = *(const uint4*)(p + 16 + half * 8);
  return a;
}
// B 32x16: lane<16 holds col K{0..7}+{16..23}, lane>=16 same col K{8..15}+{24..31}
__device__ __forceinline__ v16bf load_b_frag(const unsigned short* Bs, int col, int half) {
  v16bf b;
  const unsigned short* p = Bs + col * BK + half * 8;
  *((uint4*)&b)       = *(const uint4*)(p);
  *(((uint4*)&b) + 1) = *(const uint4*)(p + 16);
  return b;
}

// one 32-deep k-step: 8 x v_wmma_f32_16x16x32_bf16 per wave
__device__ __forceinline__ void mma_step(const unsigned short* As, const unsigned short* Bs,
                                         int wave_m, int wave_n, int lane, v8f acc[2][4]) {
  const int half = lane >> 4, l16 = lane & 15;
  v16bf a0 = load_a_frag(As, wave_m * 32 + l16, half);
  v16bf a1 = load_a_frag(As, wave_m * 32 + 16 + l16, half);
  v16bf b[4];
#pragma unroll
  for (int j = 0; j < 4; ++j) b[j] = load_b_frag(Bs, wave_n * 64 + j * 16 + l16, half);
#pragma unroll
  for (int j = 0; j < 4; ++j) {
    acc[0][j] = __builtin_amdgcn_wmma_f32_16x16x32_bf16(false, a0, false, b[j],
                                                        (short)0, acc[0][j], false, false);
    acc[1][j] = __builtin_amdgcn_wmma_f32_16x16x32_bf16(false, a1, false, b[j],
                                                        (short)0, acc[1][j], false, false);
  }
}

// ---- global -> LDS tile loaders ------------------------------------------
// bf16 tile: async DMA straight into LDS (no VGPR staging) when available
__device__ __forceinline__ void load_tile_bf16(const unsigned short* __restrict__ G, int ld,
                                               int row0, int rowMax, int k0,
                                               unsigned short* S, int tid) {
#pragma unroll
  for (int i = 0; i < 2; ++i) {
    int e = tid + i * 256;
    int r = e >> 2, q = e & 3;
    int gr = row0 + r; gr = gr > rowMax ? rowMax : gr;
    const unsigned short* p = G + (size_t)gr * ld + k0 + q * 8;
    unsigned short* s = S + r * BK + q * 8;
#if HAS_ASYNC_LDS
    __builtin_amdgcn_global_load_async_to_lds_b128(
        (gptr_b128)(__SIZE_TYPE__)p,
        (lptr_b128)(__attribute__((address_space(3))) void*)s, 0, 0);
#else
    *(uint4*)s = *(const uint4*)p;
#endif
  }
}

// f32 tile converted to bf16 on the fly (must stage through VGPRs)
__device__ __forceinline__ void load_tile_f32_bf16(const float* __restrict__ G, int ld,
                                                   int row0, int rowMax, int k0,
                                                   unsigned short* S, int tid) {
#pragma unroll
  for (int i = 0; i < 2; ++i) {
    int e = tid + i * 256;
    int r = e >> 2, q = e & 3;
    int gr = row0 + r; gr = gr > rowMax ? rowMax : gr;
    const float* p = G + (size_t)gr * ld + k0 + q * 8;
    float4 f0 = *(const float4*)(p);
    float4 f1 = *(const float4*)(p + 4);
    unsigned short* d = S + r * BK + q * 8;
    d[0] = f2bf(f0.x); d[1] = f2bf(f0.y); d[2] = f2bf(f0.z); d[3] = f2bf(f0.w);
    d[4] = f2bf(f1.x); d[5] = f2bf(f1.y); d[6] = f2bf(f1.z); d[7] = f2bf(f1.w);
  }
}

// ---- double-buffered GEMM main loop --------------------------------------
// loadA(dstLds, k0) / loadB(dstLds, k0) fill one BMxBK / BNxBK bf16 tile.
template <typename LA, typename LB>
__device__ __forceinline__ void gemm_pipeline(LA loadA, LB loadB, int K,
                                              unsigned short (*As)[BM * BK],
                                              unsigned short (*Bs)[BN * BK],
                                              int wave_m, int wave_n, int lane,
                                              v8f acc[2][4]) {
  loadA(As[0], 0);
  loadB(Bs[0], 0);
  wait_async_lds();
  __syncthreads();
  for (int k0 = 0; k0 < K; k0 += BK) {
    const int cur = (k0 / BK) & 1;
    if (k0 + BK < K) {                 // prefetch next tile into other buffer
      loadA(As[cur ^ 1], k0 + BK);
      loadB(Bs[cur ^ 1], k0 + BK);
    }
    mma_step(As[cur], Bs[cur], wave_m, wave_n, lane, acc);
    wait_async_lds();                  // own async loads landed in LDS
    __syncthreads();                   // ...and everyone's; cur buffer free
  }
}

// ---------------------------------------------------------------------------
// fp32 -> bf16 conversion
__global__ __launch_bounds__(256) void cvt_f32_bf16_kernel(const float* __restrict__ src,
                                                           unsigned short* __restrict__ dst, int n) {
  int i = blockIdx.x * 256 + threadIdx.x;
  if (i < n) dst[i] = f2bf(src[i]);
}

// ---------------------------------------------------------------------------
// Fused QKV projection: C[m, n] = x[m, :] . Wqkv[n, :] + bias, scattered to
// Qh/Kh [b,h,s,hd] and Vt [b,h,hd,s] as bf16.
__global__ __launch_bounds__(256) void qkv_gemm_kernel(
    const unsigned short* __restrict__ X, const unsigned short* __restrict__ W,
    const float* __restrict__ bq, const float* __restrict__ bk, const float* __restrict__ bv,
    unsigned short* __restrict__ Qh, unsigned short* __restrict__ Kh,
    unsigned short* __restrict__ Vt) {
  __shared__ __align__(16) unsigned short As[2][BM * BK];
  __shared__ __align__(16) unsigned short Bs[2][BN * BK];
  const int tid = threadIdx.x, lane = tid & 31, wave = tid >> 5;
  const int wave_m = wave >> 1, wave_n = wave & 1;
  const int row0 = blockIdx.y * BM, col0 = blockIdx.x * BN;
  v8f acc[2][4] = {};
  gemm_pipeline(
      [&](unsigned short* S, int k0) { load_tile_bf16(X, D_MODEL, row0, MTOT - 1, k0, S, tid); },
      [&](unsigned short* S, int k0) { load_tile_bf16(W, D_MODEL, col0, 3 * D_MODEL - 1, k0, S, tid); },
      D_MODEL, As, Bs, wave_m, wave_n, lane, acc);
  const int half = lane >> 4, l16 = lane & 15;
#pragma unroll
  for (int im = 0; im < 2; ++im)
#pragma unroll
    for (int jn = 0; jn < 4; ++jn) {
      const int n = col0 + wave_n * 64 + jn * 16 + l16;
      const int qkv = n >> 10, d = n & 1023, h = d >> 6, hd = d & 63;
      const float* bsel = (qkv == 0) ? bq : ((qkv == 1) ? bk : bv);
      const float bias = bsel[d];
#pragma unroll
      for (int r = 0; r < 8; ++r) {
        const int m = row0 + wave_m * 32 + im * 16 + r + half * 8;
        const int b = m >> 11, s = m & 2047;
        const unsigned short v = f2bf(acc[im][jn][r] + bias);
        const size_t hb = (size_t)(b * N_HEADS + h);
        if (qkv == 0)      Qh[(hb * SEQ + s) * HEAD_DIM + hd] = v;
        else if (qkv == 1) Kh[(hb * SEQ + s) * HEAD_DIM + hd] = v;
        else               Vt[(hb * HEAD_DIM + hd) * SEQ + s] = v;
      }
    }
}

// ---------------------------------------------------------------------------
// scores[z, q, k] = (Q[z,q,:] . K[z,k,:]) / sqrt(Hd)   (f32, into d_out attn)
__global__ __launch_bounds__(256) void scores_gemm_kernel(
    const unsigned short* __restrict__ Qh, const unsigned short* __restrict__ Kh,
    float* __restrict__ attn) {
  __shared__ __align__(16) unsigned short As[2][BM * BK];
  __shared__ __align__(16) unsigned short Bs[2][BN * BK];
  const int tid = threadIdx.x, lane = tid & 31, wave = tid >> 5;
  const int wave_m = wave >> 1, wave_n = wave & 1;
  const int z = blockIdx.z;
  const unsigned short* A = Qh + (size_t)z * SEQ * HEAD_DIM;
  const unsigned short* B = Kh + (size_t)z * SEQ * HEAD_DIM;
  float* C = attn + (size_t)z * SEQ * SEQ;
  const int row0 = blockIdx.y * BM, col0 = blockIdx.x * BN;
  v8f acc[2][4] = {};
  gemm_pipeline(
      [&](unsigned short* S, int k0) { load_tile_bf16(A, HEAD_DIM, row0, SEQ - 1, k0, S, tid); },
      [&](unsigned short* S, int k0) { load_tile_bf16(B, HEAD_DIM, col0, SEQ - 1, k0, S, tid); },
      HEAD_DIM, As, Bs, wave_m, wave_n, lane, acc);
  const int half = lane >> 4, l16 = lane & 15;
  const float scale = 0.125f;   // 1/sqrt(64)
#pragma unroll
  for (int im = 0; im < 2; ++im)
#pragma unroll
    for (int jn = 0; jn < 4; ++jn) {
      const int n = col0 + wave_n * 64 + jn * 16 + l16;
#pragma unroll
      for (int r = 0; r < 8; ++r) {
        const int m = row0 + wave_m * 32 + im * 16 + r + half * 8;
        C[(size_t)m * SEQ + n] = acc[im][jn][r] * scale;
      }
    }
}

// ---------------------------------------------------------------------------
// in-place row softmax over attn rows of length SEQ (one block per row)
__global__ __launch_bounds__(256) void softmax_kernel(float* __restrict__ attn) {
  __shared__ float red[256];
  const int tid = threadIdx.x;
  float* p = attn + (size_t)blockIdx.x * SEQ;
  float v[SEQ / 256];
  float mx = -3.0e38f;
#pragma unroll
  for (int i = 0; i < SEQ / 256; ++i) { v[i] = p[tid + i * 256]; mx = fmaxf(mx, v[i]); }
  red[tid] = mx; __syncthreads();
  for (int s = 128; s > 0; s >>= 1) { if (tid < s) red[tid] = fmaxf(red[tid], red[tid + s]); __syncthreads(); }
  mx = red[0]; __syncthreads();
  float sum = 0.f;
#pragma unroll
  for (int i = 0; i < SEQ / 256; ++i) { v[i] = __expf(v[i] - mx); sum += v[i]; }
  red[tid] = sum; __syncthreads();
  for (int s = 128; s > 0; s >>= 1) { if (tid < s) red[tid] += red[tid + s]; __syncthreads(); }
  const float inv = 1.0f / red[0];
#pragma unroll
  for (int i = 0; i < SEQ / 256; ++i) p[tid + i * 256] = v[i] * inv;
}

// ---------------------------------------------------------------------------
// ctx[b, s, h*64+n] = sum_k attn[z, s, k] * V[z, k, n]  (A f32->bf16 on load,
// B = V transposed [hd, s]); output bf16 in [b, s, d] layout.
__global__ __launch_bounds__(256) void ctx_gemm_kernel(
    const float* __restrict__ attn, const unsigned short* __restrict__ Vt,
    unsigned short* __restrict__ Ctx) {
  __shared__ __align__(16) unsigned short As[2][BM * BK];
  __shared__ __align__(16) unsigned short Bs[2][BN * BK];
  const int tid = threadIdx.x, lane = tid & 31, wave = tid >> 5;
  const int wave_m = wave >> 1, wave_n = wave & 1;
  const int z = blockIdx.z, b = z >> 4, h = z & 15;
  const float* A = attn + (size_t)z * SEQ * SEQ;
  const unsigned short* B = Vt + (size_t)z * HEAD_DIM * SEQ;
  const int row0 = blockIdx.y * BM;
  v8f acc[2][4] = {};
  gemm_pipeline(
      [&](unsigned short* S, int k0) { load_tile_f32_bf16(A, SEQ, row0, SEQ - 1, k0, S, tid); },
      [&](unsigned short* S, int k0) { load_tile_bf16(B, SEQ, 0, HEAD_DIM - 1, k0, S, tid); },
      SEQ, As, Bs, wave_m, wave_n, lane, acc);
  const int half = lane >> 4, l16 = lane & 15;
#pragma unroll
  for (int im = 0; im < 2; ++im)
#pragma unroll
    for (int jn = 0; jn < 4; ++jn) {
      const int n = wave_n * 64 + jn * 16 + l16;
      if (n < HEAD_DIM) {
#pragma unroll
        for (int r = 0; r < 8; ++r) {
          const int m = row0 + wave_m * 32 + im * 16 + r + half * 8;
          Ctx[((size_t)(b * SEQ + m)) * D_MODEL + h * HEAD_DIM + n] = f2bf(acc[im][jn][r]);
        }
      }
    }
}

// ---------------------------------------------------------------------------
// out[m, n] = ctx[m, :] . Wo[n, :] + bo[n]   (f32 into d_out)
__global__ __launch_bounds__(256) void out_gemm_kernel(
    const unsigned short* __restrict__ Ctx, const unsigned short* __restrict__ Wo,
    const float* __restrict__ bo, float* __restrict__ Out) {
  __shared__ __align__(16) unsigned short As[2][BM * BK];
  __shared__ __align__(16) unsigned short Bs[2][BN * BK];
  const int tid = threadIdx.x, lane = tid & 31, wave = tid >> 5;
  const int wave_m = wave >> 1, wave_n = wave & 1;
  const int row0 = blockIdx.y * BM, col0 = blockIdx.x * BN;
  v8f acc[2][4] = {};
  gemm_pipeline(
      [&](unsigned short* S, int k0) { load_tile_bf16(Ctx, D_MODEL, row0, MTOT - 1, k0, S, tid); },
      [&](unsigned short* S, int k0) { load_tile_bf16(Wo, D_MODEL, col0, D_MODEL - 1, k0, S, tid); },
      D_MODEL, As, Bs, wave_m, wave_n, lane, acc);
  const int half = lane >> 4, l16 = lane & 15;
#pragma unroll
  for (int im = 0; im < 2; ++im)
#pragma unroll
    for (int jn = 0; jn < 4; ++jn) {
      const int n = col0 + wave_n * 64 + jn * 16 + l16;
      const float bias = bo[n];
#pragma unroll
      for (int r = 0; r < 8; ++r) {
        const int m = row0 + wave_m * 32 + im * 16 + r + half * 8;
        Out[(size_t)m * D_MODEL + n] = acc[im][jn][r] + bias;
      }
    }
}

// ---------------------------------------------------------------------------
extern "C" void kernel_launch(void* const* d_in, const int* in_sizes, int n_in,
                              void* d_out, int out_size, void* d_ws, size_t ws_size,
                              hipStream_t stream) {
  const float* x  = (const float*)d_in[0];
  const float* Wq = (const float*)d_in[1];
  const float* bq = (const float*)d_in[2];
  const float* Wk = (const float*)d_in[3];
  const float* bk = (const float*)d_in[4];
  const float* Wv = (const float*)d_in[5];
  const float* bv = (const float*)d_in[6];
  const float* Wo = (const float*)d_in[7];
  const float* bo = (const float*)d_in[8];

  float* out  = (float*)d_out;
  float* attn = out + (size_t)MTOT * D_MODEL;   // [B,H,S,S] f32

  // workspace carve-up (all bf16 stored as u16); total ~48 MB
  unsigned short* x16   = (unsigned short*)d_ws;
  unsigned short* wqkv  = x16   + (size_t)MTOT * D_MODEL;          // [3072,1024]
  unsigned short* wo16  = wqkv  + (size_t)3 * D_MODEL * D_MODEL;   // [1024,1024]
  unsigned short* qh    = wo16  + (size_t)D_MODEL * D_MODEL;       // [B,H,S,Hd]
  unsigned short* kh    = qh    + (size_t)MTOT * D_MODEL;
  unsigned short* vt    = kh    + (size_t)MTOT * D_MODEL;          // [B,H,Hd,S]
  unsigned short* ctx16 = vt    + (size_t)MTOT * D_MODEL;          // [B,S,D]

  const int nx = MTOT * D_MODEL;
  const int nw = D_MODEL * D_MODEL;
  cvt_f32_bf16_kernel<<<(nx + 255) / 256, 256, 0, stream>>>(x, x16, nx);
  cvt_f32_bf16_kernel<<<(nw + 255) / 256, 256, 0, stream>>>(Wq, wqkv,          nw);
  cvt_f32_bf16_kernel<<<(nw + 255) / 256, 256, 0, stream>>>(Wk, wqkv + nw,     nw);
  cvt_f32_bf16_kernel<<<(nw + 255) / 256, 256, 0, stream>>>(Wv, wqkv + 2 * nw, nw);
  cvt_f32_bf16_kernel<<<(nw + 255) / 256, 256, 0, stream>>>(Wo, wo16,          nw);

  qkv_gemm_kernel<<<dim3(3 * D_MODEL / BN, MTOT / BM), 256, 0, stream>>>(
      x16, wqkv, bq, bk, bv, qh, kh, vt);

  scores_gemm_kernel<<<dim3(SEQ / BN, SEQ / BM, BATCH * N_HEADS), 256, 0, stream>>>(
      qh, kh, attn);

  softmax_kernel<<<BATCH * N_HEADS * SEQ, 256, 0, stream>>>(attn);

  ctx_gemm_kernel<<<dim3(1, SEQ / BM, BATCH * N_HEADS), 256, 0, stream>>>(
      attn, vt, ctx16);

  out_gemm_kernel<<<dim3(D_MODEL / BN, MTOT / BM), 256, 0, stream>>>(
      ctx16, wo16, bo, out);
}
